// SelfAttention_18760417149306
// MI455X (gfx1250) — matrix operand order repaired and checked by reading, not compiled, not run on hardware
//
#include <hip/hip_runtime.h>
#include <hip/hip_bf16.h>

// Problem constants (from reference): B=16, C=256, H=W=64 -> N=4096, D=C/8=32
#define BATCH 16
#define CDIM  256
#define NDIM  4096
#define DDIM  32

typedef __attribute__((ext_vector_type(16))) __bf16 v16bf;
typedef __attribute__((ext_vector_type(8)))  float  v8f;

union BV { v16bf v; uint4 u[2]; };

// A-operand (16x32 bf16, rows across lanes, K quarter-interleaved per ISA table):
// lanes 0-15 need K = {0..7} and {16..23}; lanes 16-31 need K = {8..15} and {24..31}.
// rowbase points at 64 contiguous bytes (32 bf16, K ascending).
__device__ __forceinline__ v16bf load_a64(const void* rowbase, int hf) {
  BV x;
  x.u[0] = *(const uint4*)((const char*)rowbase + hf * 16);
  x.u[1] = *(const uint4*)((const char*)rowbase + 32 + hf * 16);
  return x.v;
}
// B-operand (32x16 bf16, cols across lanes, K quarter-interleaved, same as A):
// lanes 0-15 hold K = {0..7} and {16..23}; lanes 16-31 hold K = {8..15} and {24..31}.
__device__ __forceinline__ v16bf load_b64(const void* colbase, int hf) {
  BV x;
  x.u[0] = *(const uint4*)((const char*)colbase + hf * 16);
  x.u[1] = *(const uint4*)((const char*)colbase + 32 + hf * 16);
  return x.v;
}
__device__ __forceinline__ v8f wmma_bf16(v16bf a, v16bf b, v8f c) {
  return __builtin_amdgcn_wmma_f32_16x16x32_bf16(false, a, false, b, (short)0, c,
                                                 false, false);
}

// ---------------- Kernel 1: x[B][C][N] f32 -> xb[B][N][C] bf16 (tiled transpose) ----
__global__ __launch_bounds__(256) void k_convert_x(const float* __restrict__ x,
                                                   __bf16* __restrict__ xb) {
  __shared__ float tile[32][33];
  const int b = blockIdx.z, c0 = blockIdx.y * 32, n0 = blockIdx.x * 32;
  const int tx = threadIdx.x, ty = threadIdx.y;  // 32 x 8
  const float* xp = x + ((size_t)b * CDIM + c0) * NDIM + n0;
#pragma unroll
  for (int i = 0; i < 4; ++i)
    tile[ty + i * 8][tx] = xp[(size_t)(ty + i * 8) * NDIM + tx];
  __syncthreads();
  __bf16* op = xb + ((size_t)b * NDIM + n0) * CDIM + c0;
#pragma unroll
  for (int i = 0; i < 4; ++i) {
    const int n = ty + i * 8;
    op[(size_t)n * CDIM + tx] = (__bf16)tile[tx][n];
  }
}

// ---------------- Kernel 1b: weights f32 -> bf16 (layouts preserved) ----------------
__global__ void k_convert_w(const float* __restrict__ Wq, const float* __restrict__ Wk,
                            const float* __restrict__ Wv, __bf16* __restrict__ wq,
                            __bf16* __restrict__ wk, __bf16* __restrict__ wv) {
  const int idx = blockIdx.x * 256 + threadIdx.x;
  const int QK = DDIM * CDIM;
  if (idx < QK)                       wq[idx] = (__bf16)Wq[idx];
  else if (idx < 2 * QK)              wk[idx - QK] = (__bf16)Wk[idx - QK];
  else if (idx < 2 * QK + CDIM*CDIM)  wv[idx - 2 * QK] = (__bf16)Wv[idx - 2 * QK];
}

// ---------------- Kernel 2: q/k projection. 1 wave per 16-row n-tile. --------------
// q[n,d] = sum_c xb[n,c] * Wq[d,c]; outputs qT/kT stored [B][N][D] bf16.
__global__ __launch_bounds__(32) void k_proj_qk(const __bf16* __restrict__ xb,
                                                const __bf16* __restrict__ wq,
                                                const __bf16* __restrict__ wk,
                                                __bf16* __restrict__ qT,
                                                __bf16* __restrict__ kT) {
  const int b = blockIdx.y, n0 = blockIdx.x * 16;
  const int lane = threadIdx.x & 31, lc = lane & 15, hf = lane >> 4;
  v8f q0 = {}, q1 = {}, k0 = {}, k1 = {};
  const __bf16* xrow = xb + ((size_t)b * NDIM + n0 + lc) * CDIM;  // A row = n
#pragma unroll
  for (int ks = 0; ks < CDIM; ks += 32) {
    const v16bf A   = load_a64(xrow + ks, hf);
    const v16bf bq0 = load_b64(wq + (size_t)lc * CDIM + ks, hf);         // col d = lc
    const v16bf bq1 = load_b64(wq + (size_t)(16 + lc) * CDIM + ks, hf);  // col d = 16+lc
    const v16bf bk0 = load_b64(wk + (size_t)lc * CDIM + ks, hf);
    const v16bf bk1 = load_b64(wk + (size_t)(16 + lc) * CDIM + ks, hf);
    q0 = wmma_bf16(A, bq0, q0);
    q1 = wmma_bf16(A, bq1, q1);
    k0 = wmma_bf16(A, bk0, k0);
    k1 = wmma_bf16(A, bk1, k1);
  }
#pragma unroll
  for (int r = 0; r < 8; ++r) {  // C-tile: row n = r + 8*hf (this lane half), col = lc
    const int n = n0 + r + 8 * hf;
    __bf16* q = qT + ((size_t)b * NDIM + n) * DDIM;
    __bf16* k = kT + ((size_t)b * NDIM + n) * DDIM;
    q[lc] = (__bf16)q0[r];  q[16 + lc] = (__bf16)q1[r];
    k[lc] = (__bf16)k0[r];  k[16 + lc] = (__bf16)k1[r];
  }
}

// ---------------- Kernel 3: v projection. 8 waves/block, wave w -> 32 channels. -----
// v[e,n] = sum_c Wv[e,c] * xb[n,c]; output vb stored [B][C][N] bf16 (n contiguous).
__global__ __launch_bounds__(256) void k_proj_v(const __bf16* __restrict__ xb,
                                                const __bf16* __restrict__ wv,
                                                __bf16* __restrict__ vb) {
  const int b = blockIdx.y, n0 = blockIdx.x * 16;
  const int w = threadIdx.x >> 5, lane = threadIdx.x & 31;
  const int lc = lane & 15, hf = lane >> 4, e0 = w * 32;
  v8f a0 = {}, a1 = {};
  const __bf16* xcol = xb + ((size_t)b * NDIM + n0 + lc) * CDIM;  // B col = n
  const __bf16* wr0  = wv + (size_t)(e0 + lc) * CDIM;             // A row = e
  const __bf16* wr1  = wv + (size_t)(e0 + 16 + lc) * CDIM;
#pragma unroll
  for (int ks = 0; ks < CDIM; ks += 32) {
    const v16bf Bx = load_b64(xcol + ks, hf);
    a0 = wmma_bf16(load_a64(wr0 + ks, hf), Bx, a0);
    a1 = wmma_bf16(load_a64(wr1 + ks, hf), Bx, a1);
  }
#pragma unroll
  for (int r = 0; r < 8; ++r) {  // rows = e, cols = n
    const int er = r + 8 * hf;
    vb[((size_t)b * CDIM + e0 + er) * NDIM + n0 + lc]      = (__bf16)a0[r];
    vb[((size_t)b * CDIM + e0 + 16 + er) * NDIM + n0 + lc] = (__bf16)a1[r];
  }
}

// ---------------- Kernel 4: flash attention -----------------------------------------
// softmax over axis n (query axis of reference) => out[:,m] = sum_n softmax_n(s[m,n]) v[:,n]
// with s[m,n] = k[:,m] . q[n,:].  Block = 8 waves, one (b, 16-row m tile).
// Per 256-n outer step: each wave produces one 32-col score chunk + stats into LDS,
// then every wave consumes all 8 chunks into its private 32-channel f32 accumulator.
__global__ __launch_bounds__(256) void k_flash(const __bf16* __restrict__ qT,
                                               const __bf16* __restrict__ kT,
                                               const __bf16* __restrict__ vb,
                                               const float* __restrict__ x,
                                               const float* __restrict__ gptr,
                                               float* __restrict__ out) {
  __shared__ float lds_P[8 * 16 * 32];  // 8 chunks of 16x32 probs (f32, row-major)
  __shared__ float lds_max[8 * 16];
  __shared__ float lds_sum[8 * 16];
  const int b = blockIdx.y, m0 = blockIdx.x * 16;
  const int w = threadIdx.x >> 5, lane = threadIdx.x & 31;
  const int lc = lane & 15, hf = lane >> 4, e0 = w * 32;
  const float gamma = gptr[0];

  // Score A-operand: rows m, K = d (D==32 -> exactly one WMMA deep). Loaded once.
  const v16bf a_k = load_a64(kT + ((size_t)b * NDIM + m0 + lc) * DDIM, hf);

  v8f o0 = {}, o1 = {};
  float run_max = -1e30f, run_sum = 0.0f;
  const v8f zero = {};

  for (int it = 0; it < NDIM / 256; ++it) {
    const int nbase = it * 256;
    const int nw = nbase + w * 32;  // this wave's produce chunk
    // ---- produce ----
    const v16bf bq0 = load_b64(qT + ((size_t)b * NDIM + nw + lc) * DDIM, hf);
    const v16bf bq1 = load_b64(qT + ((size_t)b * NDIM + nw + 16 + lc) * DDIM, hf);
    v8f s0 = wmma_bf16(a_k, bq0, zero);
    v8f s1 = wmma_bf16(a_k, bq1, zero);
    float rmax[8], rsum[8], p0[8], p1[8];
#pragma unroll
    for (int r = 0; r < 8; ++r) {
      float t = fmaxf(s0[r], s1[r]);
      t = fmaxf(t, __shfl_xor(t, 1, 16));
      t = fmaxf(t, __shfl_xor(t, 2, 16));
      t = fmaxf(t, __shfl_xor(t, 4, 16));
      t = fmaxf(t, __shfl_xor(t, 8, 16));
      rmax[r] = t;
      p0[r] = __expf(s0[r] - t);
      p1[r] = __expf(s1[r] - t);
      float s = p0[r] + p1[r];
      s += __shfl_xor(s, 1, 16);
      s += __shfl_xor(s, 2, 16);
      s += __shfl_xor(s, 4, 16);
      s += __shfl_xor(s, 8, 16);
      rsum[r] = s;
    }
    {
      float* P = &lds_P[w * 512];
#pragma unroll
      for (int r = 0; r < 8; ++r) {
        const int row = r + 8 * hf;
        P[row * 32 + lc]      = p0[r];
        P[row * 32 + 16 + lc] = p1[r];
      }
      if (lc == 0) {
#pragma unroll
        for (int r = 0; r < 8; ++r) {
          lds_max[w * 16 + r + 8 * hf] = rmax[r];
          lds_sum[w * 16 + r + 8 * hf] = rsum[r];
        }
      }
    }
    __syncthreads();
    // ---- combine stats: this lane tracks row m = lc ----
    float cmax[8], csum[8], bm = -1e30f;
#pragma unroll
    for (int j = 0; j < 8; ++j) {
      cmax[j] = lds_max[j * 16 + lc];
      csum[j] = lds_sum[j * 16 + lc];
      bm = fmaxf(bm, cmax[j]);
    }
    const float newmax = fmaxf(run_max, bm);
    const float scale_old = __expf(run_max - newmax);
    float alpha[8], add = 0.0f;
#pragma unroll
    for (int j = 0; j < 8; ++j) {
      alpha[j] = __expf(cmax[j] - newmax);
      add += alpha[j] * csum[j];
    }
    run_sum = run_sum * scale_old + add;
    run_max = newmax;
#pragma unroll
    for (int r = 0; r < 8; ++r) {  // rescale accumulators, per row r + 8*hf
      const float sc = __shfl(scale_old, r + 8 * hf);
      o0[r] *= sc;
      o1[r] *= sc;
    }
    // ---- consume all 8 chunks ----
    const char* vb0 = (const char*)(vb + ((size_t)b * CDIM + e0 + lc) * NDIM + nbase);
    const char* vb1 = (const char*)(vb + ((size_t)b * CDIM + e0 + 16 + lc) * NDIM + nbase);
#pragma unroll
    for (int j = 0; j < 8; ++j) {
      // A = P chunk j (rows m across lanes), scaled by exp(chunkmax - run_max)
      const float* P = &lds_P[j * 512 + lc * 32 + hf * 8];
      const float4 f0 = *(const float4*)(P);
      const float4 f1 = *(const float4*)(P + 4);
      const float4 f2 = *(const float4*)(P + 16);
      const float4 f3 = *(const float4*)(P + 20);
      const float aj = alpha[j];
      v16bf av;
      av[0]  = (__bf16)(f0.x * aj); av[1]  = (__bf16)(f0.y * aj);
      av[2]  = (__bf16)(f0.z * aj); av[3]  = (__bf16)(f0.w * aj);
      av[4]  = (__bf16)(f1.x * aj); av[5]  = (__bf16)(f1.y * aj);
      av[6]  = (__bf16)(f1.z * aj); av[7]  = (__bf16)(f1.w * aj);
      av[8]  = (__bf16)(f2.x * aj); av[9]  = (__bf16)(f2.y * aj);
      av[10] = (__bf16)(f2.z * aj); av[11] = (__bf16)(f2.w * aj);
      av[12] = (__bf16)(f3.x * aj); av[13] = (__bf16)(f3.y * aj);
      av[14] = (__bf16)(f3.z * aj); av[15] = (__bf16)(f3.w * aj);
      const v16bf bv0 = load_b64(vb0 + j * 64, hf);  // B[n,e] = v[e,n], K = n
      const v16bf bv1 = load_b64(vb1 + j * 64, hf);
      o0 = wmma_bf16(av, bv0, o0);
      o1 = wmma_bf16(av, bv1, o1);
    }
    __syncthreads();  // protect lds_P before next produce
  }

  // ---- epilogue: normalize, gamma * out + x ----
  const float inv = 1.0f / run_sum;  // valid for row m = lc
#pragma unroll
  for (int t = 0; t < 2; ++t) {
    const v8f o = t ? o1 : o0;
    const int e = e0 + t * 16 + lc;
    const size_t base = ((size_t)b * CDIM + e) * NDIM + m0 + 8 * hf;
    const float* xi = x + base;
    float* oo = out + base;
    float res[8];
#pragma unroll
    for (int r = 0; r < 8; ++r) {
      const float iv = __shfl(inv, r + 8 * hf);
      res[r] = gamma * (o[r] * iv) + xi[r];
    }
    *(float4*)(oo)     = make_float4(res[0], res[1], res[2], res[3]);
    *(float4*)(oo + 4) = make_float4(res[4], res[5], res[6], res[7]);
  }
}

// ------------------------------------------------------------------------------------
extern "C" void kernel_launch(void* const* d_in, const int* in_sizes, int n_in,
                              void* d_out, int out_size, void* d_ws, size_t ws_size,
                              hipStream_t stream) {
  (void)in_sizes; (void)n_in; (void)out_size; (void)ws_size;
  const float* x  = (const float*)d_in[0];
  const float* Wq = (const float*)d_in[1];
  const float* Wk = (const float*)d_in[2];
  const float* Wv = (const float*)d_in[3];
  const float* g  = (const float*)d_in[4];
  float* out = (float*)d_out;

  char* ws = (char*)d_ws;                      // ~75.7 MB used, all L2-resident
  __bf16* xb  = (__bf16*)(ws);                 // [B][N][C]  32 MB
  __bf16* vb  = (__bf16*)(ws + 33554432);      // [B][C][N]  32 MB
  __bf16* qT  = (__bf16*)(ws + 67108864);      // [B][N][D]   4 MB
  __bf16* kT  = (__bf16*)(ws + 71303168);      // [B][N][D]   4 MB
  __bf16* wqb = (__bf16*)(ws + 75497472);      // [D][C]     16 KB
  __bf16* wkb = (__bf16*)(ws + 75513856);      // [D][C]     16 KB
  __bf16* wvb = (__bf16*)(ws + 75530240);      // [C][C]    128 KB

  k_convert_x<<<dim3(NDIM / 32, CDIM / 32, BATCH), dim3(32, 8), 0, stream>>>(x, xb);
  k_convert_w<<<(2 * DDIM * CDIM + CDIM * CDIM + 255) / 256, 256, 0, stream>>>(
      Wq, Wk, Wv, wqb, wkb, wvb);
  k_proj_qk<<<dim3(NDIM / 16, BATCH), 32, 0, stream>>>(xb, wqb, wkb, qT, kT);
  k_proj_v<<<dim3(NDIM / 16, BATCH), 256, 0, stream>>>(xb, wvb, vb);
  k_flash<<<dim3(NDIM / 16, BATCH), 256, 0, stream>>>(qT, kT, vb, x, g, out);
}
